// EMA_Codebook_83614423318835
// MI455X (gfx1250) — compile-verified, matrix operand order checked by reading, not run on hardware
//
#include <hip/hip_runtime.h>
#include <hip/hip_bf16.h>

// Problem constants (from reference)
#define N_PTS   32768
#define K_CODES 8192
#define D_DIM   256
#define DECAY_F 0.99f
#define OMD_F   0.01f      // 1 - DECAY
#define EPS_F   1e-5f

typedef __attribute__((ext_vector_type(16))) __bf16 v16bf;
typedef __attribute__((ext_vector_type(8)))  float  v8f;

// ---------------------------------------------------------------------------
// Kernel 1: c2[k] = sum_d codebook[k][d]^2   (one wave per code)
// ---------------------------------------------------------------------------
__global__ __launch_bounds__(256) void c2_kernel(const float* __restrict__ cb,
                                                 float* __restrict__ c2) {
  const int wave = threadIdx.x >> 5;
  const int lane = threadIdx.x & 31;
  const int k = blockIdx.x * 8 + wave;
  const float* row = cb + (size_t)k * D_DIM;
  float s = 0.f;
#pragma unroll
  for (int j = 0; j < 8; ++j) { float v = row[lane + 32 * j]; s += v * v; }
#pragma unroll
  for (int off = 16; off >= 1; off >>= 1) s += __shfl_xor(s, off, 32);
  if (lane == 0) c2[k] = s;
}

// ---------------------------------------------------------------------------
// Kernel 2: split codebook into bf16 hi/lo pair (one-time, enables pure-bf16
// hot loop with ~fp32-accurate 3-term product)
// ---------------------------------------------------------------------------
__global__ __launch_bounds__(256) void split_cb_kernel(const float* __restrict__ cb,
                                                       __bf16* __restrict__ cb_hi,
                                                       __bf16* __restrict__ cb_lo) {
  size_t i = (size_t)blockIdx.x * 256 + threadIdx.x;
  float v = cb[i];
  __bf16 h = (__bf16)v;
  cb_hi[i] = h;
  cb_lo[i] = (__bf16)(v - (float)h);
}

// ---------------------------------------------------------------------------
// Kernel 3: fused  argmin_k( c2[k] - 2 * z[n] . c[k] )  via bf16 WMMA.
// Block = 8 waves = 128 rows. Wave = 16 rows x (2 x 16-col subtiles / tile).
// A fragments (16 rows x 256 K, hi+lo) live in VGPRs for the whole K sweep.
// B tiles (32 codes x 256 K, hi+lo bf16): DOUBLE-BUFFERED in LDS, filled by
// the CDNA5 async copy engine (global_load_async_to_lds_b128 / ASYNCcnt) so
// staging of tile t+1 overlaps the 48 WMMAs of tile t.
// ---------------------------------------------------------------------------
#define BPAD 8                      // bf16 pad -> 16B-aligned, conflict-spread rows
#define BSTRIDE (D_DIM + BPAD)      // 264 bf16 per LDS row
#define NTILES (K_CODES / 32)

// Low 32 bits of a generic (flat) pointer to LDS == LDS byte address (ISA 10.2)
#define LDS_ADDR32(p) ((unsigned)(uintptr_t)(p))

__device__ __forceinline__ void async_copy64B(unsigned lds_dst, const void* gsrc) {
  // Copies 64 bytes per lane: IOFFSET is added to BOTH lds and global address
  // (ISA 15.18.3 GLOBAL_LOAD_ASYNC_TO_LDS pseudocode).
  asm volatile(
      "global_load_async_to_lds_b128 %0, %1, off\n\t"
      "global_load_async_to_lds_b128 %0, %1, off offset:16\n\t"
      "global_load_async_to_lds_b128 %0, %1, off offset:32\n\t"
      "global_load_async_to_lds_b128 %0, %1, off offset:48"
      :: "v"(lds_dst), "v"(gsrc) : "memory");
}

__device__ __forceinline__ void wait_asynccnt0() {
  asm volatile("s_wait_asynccnt 0x0" ::: "memory");
}

__global__ __launch_bounds__(256)
void argmin_kernel(const float*  __restrict__ z,
                   const __bf16* __restrict__ cb_hi,
                   const __bf16* __restrict__ cb_lo,
                   const float*  __restrict__ c2g,
                   int*          __restrict__ idx_out,
                   float*        __restrict__ idx_out_f) {
  __shared__ __bf16 BsHi[2][32 * BSTRIDE];
  __shared__ __bf16 BsLo[2][32 * BSTRIDE];

  const int tid = threadIdx.x;
  const int wave = tid >> 5;
  const int lane = tid & 31;
  const int row_lane = lane & 15;     // A row within wave tile / B column
  const int kgrp = lane >> 4;         // which 16-wide K half this lane holds

  // staging assignment: thread stages 64B (32 bf16) of one code row per matrix
  const int cload = tid >> 3;          // 0..31 code slot
  const int coff  = (tid & 7) * 32;    // 0..224 element offset
  const unsigned ldsH0 = LDS_ADDR32(&BsHi[0][cload * BSTRIDE + coff]);
  const unsigned ldsH1 = LDS_ADDR32(&BsHi[1][cload * BSTRIDE + coff]);
  const unsigned ldsL0 = LDS_ADDR32(&BsLo[0][cload * BSTRIDE + coff]);
  const unsigned ldsL1 = LDS_ADDR32(&BsLo[1][cload * BSTRIDE + coff]);

  // ---- load A fragments once: 16 rows x 256 K, split to bf16 hi/lo --------
  v16bf a_hi[8], a_lo[8];
  {
    const float* zr = z + (size_t)(blockIdx.x * 128 + wave * 16 + row_lane) * D_DIM
                        + kgrp * 16;
#pragma unroll
    for (int s = 0; s < 8; ++s) {
      const float* p = zr + s * 32;
#pragma unroll
      for (int j = 0; j < 16; ++j) {
        float v = p[j];
        __bf16 h = (__bf16)v;
        a_hi[s][j] = h;
        a_lo[s][j] = (__bf16)(v - (float)h);
      }
    }
  }

  float minv[8];
  int   mini[8];
#pragma unroll
  for (int i = 0; i < 8; ++i) { minv[i] = 3.402823466e38f; mini[i] = 0; }

  // ---- prologue: async-stage tile 0 into buffer 0 -------------------------
  {
    const size_t g = (size_t)cload * D_DIM + coff;
    async_copy64B(ldsH0, cb_hi + g);
    async_copy64B(ldsL0, cb_lo + g);
  }
  wait_asynccnt0();
  __syncthreads();

  for (int t = 0; t < NTILES; ++t) {
    const int buf = t & 1;
    // ---- async-stage tile t+1 into the spare buffer (overlaps WMMAs) ------
    if (t + 1 < NTILES) {
      const size_t g = (size_t)((t + 1) * 32 + cload) * D_DIM + coff;
      async_copy64B(buf ? ldsH0 : ldsH1, cb_hi + g);
      async_copy64B(buf ? ldsL0 : ldsL1, cb_lo + g);
    }

    // per-lane c2 scalars for this tile (L2-resident; hides under WMMAs)
    const float c20 = c2g[t * 32 + row_lane];
    const float c21 = c2g[t * 32 + 16 + row_lane];

    v8f acc0 = {};   // cols t*32 + 0..15
    v8f acc1 = {};   // cols t*32 + 16..31
#pragma unroll
    for (int s = 0; s < 8; ++s) {
      const int kofs = s * 32 + kgrp * 16;
      v16bf bh0 = *(const v16bf*)(&BsHi[buf][(row_lane)      * BSTRIDE + kofs]);
      v16bf bl0 = *(const v16bf*)(&BsLo[buf][(row_lane)      * BSTRIDE + kofs]);
      v16bf bh1 = *(const v16bf*)(&BsHi[buf][(16 + row_lane) * BSTRIDE + kofs]);
      v16bf bl1 = *(const v16bf*)(&BsLo[buf][(16 + row_lane) * BSTRIDE + kofs]);
      // 3-term split product: hi*hi + hi*lo + lo*hi  (~fp32-accurate)
      acc0 = __builtin_amdgcn_wmma_f32_16x16x32_bf16(false, a_hi[s], false, bh0, (short)0, acc0, false, false);
      acc0 = __builtin_amdgcn_wmma_f32_16x16x32_bf16(false, a_hi[s], false, bl0, (short)0, acc0, false, false);
      acc0 = __builtin_amdgcn_wmma_f32_16x16x32_bf16(false, a_lo[s], false, bh0, (short)0, acc0, false, false);
      acc1 = __builtin_amdgcn_wmma_f32_16x16x32_bf16(false, a_hi[s], false, bh1, (short)0, acc1, false, false);
      acc1 = __builtin_amdgcn_wmma_f32_16x16x32_bf16(false, a_hi[s], false, bl1, (short)0, acc1, false, false);
      acc1 = __builtin_amdgcn_wmma_f32_16x16x32_bf16(false, a_lo[s], false, bh1, (short)0, acc1, false, false);
    }

    // epilogue: score = c2 - 2*dot ; update running min/argmin in registers
    const int col0 = t * 32 + row_lane;
    const int col1 = col0 + 16;
#pragma unroll
    for (int i = 0; i < 8; ++i) {
      float s0 = c20 - 2.0f * acc0[i];
      if (s0 < minv[i]) { minv[i] = s0; mini[i] = col0; }
      float s1 = c21 - 2.0f * acc1[i];
      if (s1 < minv[i]) { minv[i] = s1; mini[i] = col1; }
    }

    // make tile t+1 visible to all waves before the next compute phase
    wait_asynccnt0();
    __syncthreads();
  }

  // cross-lane argmin within each 16-lane half (C layout: VGPR i holds
  // row i for lanes 0-15 and row i+8 for lanes 16-31)
#pragma unroll
  for (int i = 0; i < 8; ++i) {
    float v = minv[i];
    int   ix = mini[i];
#pragma unroll
    for (int off = 8; off >= 1; off >>= 1) {
      float v2 = __shfl_xor(v, off, 32);
      int  ix2 = __shfl_xor(ix, off, 32);
      if (v2 < v || (v2 == v && ix2 < ix)) { v = v2; ix = ix2; }
    }
    if (row_lane == 0) {
      int r = blockIdx.x * 128 + wave * 16 + i + 8 * kgrp;
      idx_out[r] = ix;
      idx_out_f[r] = (float)ix;
    }
  }
}

// ---------------------------------------------------------------------------
// Kernel 4: EMA init: out = old * DECAY ; zero the n accumulator
// ---------------------------------------------------------------------------
__global__ __launch_bounds__(256)
void init_ema_kernel(const float* __restrict__ emb_avg, const float* __restrict__ cs,
                     float* __restrict__ out_emb, float* __restrict__ out_cs,
                     float* __restrict__ n_total) {
  size_t i = (size_t)blockIdx.x * 256 + threadIdx.x;
  out_emb[i] = emb_avg[i] * DECAY_F;
  if (i < K_CODES) out_cs[i] = cs[i] * DECAY_F;
  if (i == 0) *n_total = 0.0f;
}

// ---------------------------------------------------------------------------
// Kernel 5: gather quantized + scatter-add EMA contributions (segment sums)
// ---------------------------------------------------------------------------
__global__ __launch_bounds__(256)
void scatter_kernel(const float* __restrict__ z, const float* __restrict__ cb,
                    const int* __restrict__ idx, float* __restrict__ quant,
                    float* __restrict__ out_emb, float* __restrict__ out_cs) {
  const int n = blockIdx.x;
  const int d = threadIdx.x;
  const int k = idx[n];
  const size_t zi = (size_t)n * D_DIM + d;
  quant[zi] = cb[(size_t)k * D_DIM + d];
  atomicAdd(&out_emb[(size_t)k * D_DIM + d], z[zi] * OMD_F);
  if (d == 0) atomicAdd(&out_cs[k], OMD_F);
}

// ---------------------------------------------------------------------------
// Kernel 6: n = sum(new_cluster_size)
// ---------------------------------------------------------------------------
__global__ __launch_bounds__(256)
void reduce_n_kernel(const float* __restrict__ out_cs, float* __restrict__ n_total) {
  float v = out_cs[blockIdx.x * 256 + threadIdx.x];
#pragma unroll
  for (int off = 16; off >= 1; off >>= 1) v += __shfl_xor(v, off, 32);
  __shared__ float wsum[8];
  if ((threadIdx.x & 31) == 0) wsum[threadIdx.x >> 5] = v;
  __syncthreads();
  if (threadIdx.x == 0) {
    float s = 0.f;
#pragma unroll
    for (int w = 0; w < 8; ++w) s += wsum[w];
    atomicAdd(n_total, s);
  }
}

// ---------------------------------------------------------------------------
// Kernel 7: Laplace smoothing + codebook divide
// ---------------------------------------------------------------------------
__global__ __launch_bounds__(256)
void finalize_kernel(const float* __restrict__ out_emb, const float* __restrict__ out_cs,
                     const float* __restrict__ n_total, float* __restrict__ out_cb) {
  const int k = blockIdx.x;
  const int d = threadIdx.x;
  const float n = *n_total;
  const float smoothed = (out_cs[k] + EPS_F) / (n + (float)K_CODES * EPS_F) * n;
  const size_t i = (size_t)k * D_DIM + d;
  out_cb[i] = out_emb[i] / smoothed;
}

// ---------------------------------------------------------------------------
extern "C" void kernel_launch(void* const* d_in, const int* in_sizes, int n_in,
                              void* d_out, int out_size, void* d_ws, size_t ws_size,
                              hipStream_t stream) {
  const float* z  = (const float*)d_in[0];   // (N, D)
  const float* cb = (const float*)d_in[1];   // (K, D)
  const float* ea = (const float*)d_in[2];   // (K, D)
  const float* cs = (const float*)d_in[3];   // (K,)

  // Outputs concatenated flat in return order (all written as float):
  float* out      = (float*)d_out;
  float* quant    = out;                                  // N*D
  float* idx_f    = quant + (size_t)N_PTS * D_DIM;        // N
  float* out_cb   = idx_f + N_PTS;                        // K*D
  float* out_emb  = out_cb + (size_t)K_CODES * D_DIM;     // K*D
  float* out_cs   = out_emb + (size_t)K_CODES * D_DIM;    // K

  // Workspace layout (~8.4 MB): bf16 hi/lo codebook, c2, int indices, n.
  __bf16* cb_hi = (__bf16*)d_ws;                              // K*D
  __bf16* cb_lo = cb_hi + (size_t)K_CODES * D_DIM;            // K*D
  float*  c2    = (float*)(cb_lo + (size_t)K_CODES * D_DIM);  // K
  int*    idx_i = (int*)(c2 + K_CODES);                       // N
  float*  n_tot = (float*)(idx_i + N_PTS);                    // 1

  c2_kernel      <<<K_CODES / 8, 256, 0, stream>>>(cb, c2);
  split_cb_kernel<<<(K_CODES * D_DIM) / 256, 256, 0, stream>>>(cb, cb_hi, cb_lo);
  init_ema_kernel<<<(K_CODES * D_DIM) / 256, 256, 0, stream>>>(ea, cs, out_emb, out_cs, n_tot);
  argmin_kernel  <<<N_PTS / 128, 256, 0, stream>>>(z, cb_hi, cb_lo, c2, idx_i, idx_f);
  scatter_kernel <<<N_PTS, 256, 0, stream>>>(z, cb, idx_i, quant, out_emb, out_cs);
  reduce_n_kernel<<<K_CODES / 256, 256, 0, stream>>>(out_cs, n_tot);
  finalize_kernel<<<K_CODES, 256, 0, stream>>>(out_emb, out_cs, n_tot, out_cb);
}
